// Rnn_36945308680506
// MI455X (gfx1250) — matrix operand order repaired and checked by reading, not compiled, not run on hardware
//
#include <hip/hip_runtime.h>

typedef __attribute__((ext_vector_type(16))) _Float16 v16h;
typedef __attribute__((ext_vector_type(8)))  float    v8f;
typedef __attribute__((ext_vector_type(8)))  unsigned v8u;

#define B_TOTAL 512
#define NSIG    3
#define T_LEN   4096
#define H       17

// Branch-free tanh: t = sign(x) * (1 - 2/(exp(2|x|)+1)).
// exp2 hw op + rcp hw op; overflow -> inf -> rcp 0 -> t = 1 (correct saturation).
static __device__ __forceinline__ float fast_tanh(float x) {
    const float LOG2E_X2 = 2.88539008177792681472f;   // 2*log2(e)
    float a = __builtin_fabsf(x);
    float e = __builtin_amdgcn_exp2f(a * LOG2E_X2);   // exp(2a)
    float r = __builtin_amdgcn_rcpf(e + 1.0f);
    float t = __builtin_fmaf(-2.0f, r, 1.0f);
    return __builtin_copysignf(t, x);
}

// pack two f32 into one 32-bit word of two f16 (low = a, high = b)
static __device__ __forceinline__ unsigned pack2h(float a, float b) {
    return __builtin_bit_cast(unsigned, __builtin_amdgcn_cvt_pkrtz(a, b));
}

// One wave (32 threads) per (direction, 16-batch group).
// grid.x = 64 : blocks 0..31 -> forward, 32..63 -> backward.
// Per step: h_next(32 x 16batch) = tanh( pre + Wp(32x32) @ h ), done as two
// v_wmma_f32_16x16x32_f16 (tile0 = hidden rows 0..15, tile1 = row 16, rest 0).
__global__ __launch_bounds__(32) void rnn_scan_kernel(
    const float* __restrict__ x,
    const float* __restrict__ W_ih_f, const float* __restrict__ W_hh_f,
    const float* __restrict__ b_ih_f, const float* __restrict__ b_hh_f,
    const float* __restrict__ W_ih_b, const float* __restrict__ W_hh_b,
    const float* __restrict__ b_ih_b, const float* __restrict__ b_hh_b,
    float* __restrict__ hsum)        // [2][512][32] f32
{
    const int lane = threadIdx.x;    // 0..31
    const int half = lane >> 4;      // 0 or 1
    const int m    = lane & 15;
    const int dir  = blockIdx.x >> 5;
    const int bg   = blockIdx.x & 31;
    const int b0   = bg * 16;

    const float* W_ih = dir ? W_ih_b : W_ih_f;
    const float* W_hh = dir ? W_hh_b : W_hh_f;
    const float* b_ih = dir ? b_ih_b : b_ih_f;
    const float* b_hh = dir ? b_hh_b : b_hh_f;

    // ---- A tiles: W_hh padded to 32x32, f16, contiguous k order (B uses the same).
    // half 0: elem e -> K = e       (K in 0..15)
    // half 1: elem e -> K = e + 16  (K in 16..31)
    v16h A0 = {};  // rows M = m        (hidden-out 0..15)
    v16h A1 = {};  // rows M = 16 + m   (only m==0 -> hidden-out 16)
    #pragma unroll
    for (int e = 0; e < 16; ++e) {
        int K = e + half * 16;
        float w0 = (K < H) ? W_hh[m * H + K] : 0.0f;
        float w1 = (m == 0 && K < H) ? W_hh[16 * H + K] : 0.0f;
        A0[e] = (_Float16)w0;
        A1[e] = (_Float16)w1;
    }

    // ---- per-lane input-projection coefficients for C/D rows i = r + 8*half
    float bias0[8], wih0a[8], wih0b[8], wih0c[8];
    #pragma unroll
    for (int r = 0; r < 8; ++r) {
        int i = r + 8 * half;        // 0..15 (always a valid row)
        bias0[r] = b_ih[i] + b_hh[i];
        wih0a[r] = W_ih[i * NSIG + 0];
        wih0b[r] = W_ih[i * NSIG + 1];
        wih0c[r] = W_ih[i * NSIG + 2];
    }
    const float bias16 = b_ih[16] + b_hh[16];
    const float w16a = W_ih[16 * NSIG + 0];
    const float w16b = W_ih[16 * NSIG + 1];
    const float w16c = W_ih[16 * NSIG + 2];

    // batch column handled by this lane (same for both halves)
    const int b = b0 + m;
    const float* xb = x + (size_t)b * NSIG * T_LEN + (dir ? (T_LEN - 1) : 0);
    const int tstep = dir ? -1 : 1;

    v16h hB = {};                    // h0 = 0, f16 B layout
    v8f  s0 = {};                    // time-sums for rows i = r + 8*half
    float s16 = 0.0f;                // time-sum for row 16 (half-0 lanes)

    const float* xp = xb;
    #pragma unroll 2
    for (int s = 0; s < T_LEN; ++s) {
        const float x0 = xp[0];
        const float x1 = xp[T_LEN];
        const float x2 = xp[2 * T_LEN];
        xp += tstep;

        // C = input projection (f32)
        v8f c0, c1 = {};
        #pragma unroll
        for (int r = 0; r < 8; ++r)
            c0[r] = __builtin_fmaf(wih0c[r], x2,
                    __builtin_fmaf(wih0b[r], x1,
                    __builtin_fmaf(wih0a[r], x0, bias0[r])));
        const float pre16 = __builtin_fmaf(w16c, x2,
                            __builtin_fmaf(w16b, x1,
                            __builtin_fmaf(w16a, x0, bias16)));
        c1[0] = (half == 0) ? pre16 : 0.0f;   // tile1 row i=16 only

        // D = Wp @ h + pre
        v8f d0 = __builtin_amdgcn_wmma_f32_16x16x32_f16(
            false, A0, false, hB, (short)0, c0, false, false);
        v8f d1 = __builtin_amdgcn_wmma_f32_16x16x32_f16(
            false, A1, false, hB, (short)0, c1, false, false);

        // branch-free tanh + accumulate time-sums
        float t0[8];
        #pragma unroll
        for (int r = 0; r < 8; ++r) {
            t0[r] = fast_tanh(d0[r]);
            s0[r] += t0[r];
        }
        const float t16 = fast_tanh(d1[0]);   // == 0 on half 1 (padded rows)
        s16 += t16;

        // pack to f16 pairs (these ARE the B-layout words), then one
        // half-wave exchange of 5 words.
        unsigned pk[4];
        #pragma unroll
        for (int j = 0; j < 4; ++j) pk[j] = pack2h(t0[2 * j], t0[2 * j + 1]);
        const unsigned pk16 = pack2h(t16, 0.0f);     // low = h[16], high = 0

        unsigned ppk[4];
        #pragma unroll
        for (int j = 0; j < 4; ++j) ppk[j] = __shfl_xor(pk[j], 16, 32);
        const unsigned pp16 = __shfl_xor(pk16, 16, 32);

        // assemble next-step B operand:
        //  half 0 (cols, K=0..15):  w0..3 = own h[0..7] pairs, w4..7 = partner h[8..15]
        //  half 1 (cols, K=16..31): w0 = partner h[16] (hi 0), w1..7 = 0
        v8u hw;
        hw[0] = half ? pp16 : pk[0];
        hw[1] = half ? 0u   : pk[1];
        hw[2] = half ? 0u   : pk[2];
        hw[3] = half ? 0u   : pk[3];
        #pragma unroll
        for (int j = 0; j < 4; ++j) hw[4 + j] = half ? 0u : ppk[j];
        hB = __builtin_bit_cast(v16h, hw);
    }

    // ---- write time-sums: hsum[dir][b][i]
    float* outp = hsum + ((size_t)dir * B_TOTAL + b) * 32;
    #pragma unroll
    for (int r = 0; r < 8; ++r) outp[r + 8 * half] = s0[r];   // rows 0..15
    if (half == 0) outp[16] = s16;                            // row 16
}

// out[b][o] = conv_b[o] + (1/T) * sum_i conv_w[o][i]*fwdsum + conv_w[o][H+i]*bwdsum
__global__ void rnn_finalize_kernel(const float* __restrict__ hsum,
                                    const float* __restrict__ conv_w,
                                    const float* __restrict__ conv_b,
                                    float* __restrict__ out)
{
    const int idx = blockIdx.x * blockDim.x + threadIdx.x;
    if (idx >= B_TOTAL * 2) return;
    const int b = idx >> 1;
    const int o = idx & 1;
    const float invT = 1.0f / (float)T_LEN;
    const float* sf = hsum + (size_t)b * 32;
    const float* sb = hsum + ((size_t)B_TOTAL + b) * 32;
    float acc = conv_b[o];
    #pragma unroll
    for (int i = 0; i < H; ++i) {
        acc += conv_w[o * (2 * H) + i]     * (sf[i] * invT);
        acc += conv_w[o * (2 * H) + H + i] * (sb[i] * invT);
    }
    out[b * 2 + o] = acc;
}

extern "C" void kernel_launch(void* const* d_in, const int* in_sizes, int n_in,
                              void* d_out, int out_size, void* d_ws, size_t ws_size,
                              hipStream_t stream) {
    const float* x      = (const float*)d_in[0];
    const float* W_ih_f = (const float*)d_in[1];
    const float* W_hh_f = (const float*)d_in[2];
    const float* b_ih_f = (const float*)d_in[3];
    const float* b_hh_f = (const float*)d_in[4];
    const float* W_ih_b = (const float*)d_in[5];
    const float* W_hh_b = (const float*)d_in[6];
    const float* b_ih_b = (const float*)d_in[7];
    const float* b_hh_b = (const float*)d_in[8];
    const float* conv_w = (const float*)d_in[9];
    const float* conv_b = (const float*)d_in[10];

    float* hsum = (float*)d_ws;   // needs 2*512*32*4 = 128 KB

    rnn_scan_kernel<<<64, 32, 0, stream>>>(x,
        W_ih_f, W_hh_f, b_ih_f, b_hh_f,
        W_ih_b, W_hh_b, b_ih_b, b_hh_b, hsum);

    rnn_finalize_kernel<<<(B_TOTAL * 2 + 255) / 256, 256, 0, stream>>>(
        hsum, conv_w, conv_b, (float*)d_out);
}
